// TemporalGraphAttention_79791902425641
// MI455X (gfx1250) — matrix operand order repaired and checked by reading, not compiled, not run on hardware
//
#include <hip/hip_runtime.h>
#include <hip/hip_bf16.h>
#include <climits>
#include <stdint.h>

typedef __bf16 bf16_t;
typedef __attribute__((ext_vector_type(16))) __bf16 v16bf;
typedef __attribute__((ext_vector_type(8)))  __bf16 v8bf;
typedef __attribute__((ext_vector_type(8)))  float  v8f;

#define INDIM 256
#define TDIM  32
#define KIN   288     // IN + T
#define HC    256     // H * C
#define CDIM  64

// ---------- helpers ----------
__device__ __forceinline__ int fenc(float f) {
  int i = __float_as_int(f);
  return (i >= 0) ? i : (i ^ 0x7FFFFFFF);   // order-preserving float->int
}
__device__ __forceinline__ float fdec(int i) {
  return __int_as_float((i >= 0) ? i : (i ^ 0x7FFFFFFF));
}
__device__ __forceinline__ float elu1(float v) {
  return v > 0.f ? v : (__expf(v) - 1.f);
}

// ---------- weight prep: f32 [K x Ncols] row-major -> bf16 N-major [Ncols x K] ----------
__global__ void k_wt_prep(const float* __restrict__ W, bf16_t* __restrict__ Bt,
                          int K, int Ncols) {
  int idx = blockIdx.x * blockDim.x + threadIdx.x;
  if (idx >= K * Ncols) return;
  int k = idx / Ncols, n = idx - k * Ncols;
  Bt[(size_t)n * K + k] = (bf16_t)W[idx];
}

// ---------- build A0 = bf16[concat(x, temporal)] , one wave per node ----------
__global__ void k_build_A0(const float* __restrict__ x, const float* __restrict__ ts,
                           const float* __restrict__ W1, const float* __restrict__ b1,
                           const float* __restrict__ W2, const float* __restrict__ b2,
                           bf16_t* __restrict__ A, int N) {
  int wave = threadIdx.x >> 5, lane = threadIdx.x & 31;
  int n = blockIdx.x * 8 + wave;
  if (n >= N) return;
  for (int k = lane; k < INDIM; k += 32)
    A[(size_t)n * KIN + k] = (bf16_t)x[(size_t)n * INDIM + k];
  // temporal: t_j = b2[j] + sum_m relu(ts*W1[m]+b1[m]) * W2[m][j]; lane == j
  float t0 = ts[n];
  float acc = b2[lane];
  for (int m = 0; m < TDIM; ++m) {
    float r = fmaf(t0, W1[m], b1[m]);
    r = r > 0.f ? r : 0.f;
    acc = fmaf(r, W2[m * TDIM + lane], acc);
  }
  A[(size_t)n * KIN + INDIM + lane] = (bf16_t)acc;
}

__global__ void k_pad_bf16(bf16_t* __restrict__ A, size_t startElem, size_t count) {
  size_t idx = (size_t)blockIdx.x * blockDim.x + threadIdx.x;
  if (idx < count) A[startElem + idx] = (bf16_t)0.f;
}

// ---------- WMMA GEMM: C[f32, rows x 256] = A[bf16, rows x K] * Bt[bf16, 256 x K]^T ----------
// block = 256 threads = 8 waves; block tile 64x256; wave tile 16x128 (8 accs).
// B tile double-buffered in LDS; staged with async global->LDS copies when available.
#if defined(__gfx1250__) && \
    __has_builtin(__builtin_amdgcn_global_load_async_to_lds_b128) && \
    __has_builtin(__builtin_amdgcn_s_wait_asynccnt)
#define USE_ASYNC_LDS 1
#else
#define USE_ASYNC_LDS 0
#endif

typedef int b128_t __attribute__((vector_size(16)));   // payload type the builtin expects

__global__ __launch_bounds__(256)
void k_gemm_wmma(const bf16_t* __restrict__ A, const bf16_t* __restrict__ Bt,
                 float* __restrict__ Cf, int K, int Nrows) {
  __shared__ __align__(32) bf16_t Bs[2][HC * 32];   // 2 x 16 KB, double buffered
  const int lane = threadIdx.x & 31;
  const int wave = threadIdx.x >> 5;
  const int m0   = blockIdx.x * 64 + 16 * (wave & 3);
  const int n0   = 128 * (wave >> 2);
  const int hi   = lane >> 4;                  // 0: lanes 0-15, 1: lanes 16-31
  const int l15  = lane & 15;

  v8f acc[8];
  for (int t = 0; t < 8; ++t)
    for (int r = 0; r < 8; ++r) acc[t][r] = 0.f;

  const int arow = m0 + l15;
  const int aklo = hi * 16;                    // K-offset per A-fragment layout

  // cooperative stage of one 256x32 B tile: thread tid copies 64 B of column tid
  auto stage = [&](int kk, int buf) {
    const bf16_t* src = Bt + (size_t)threadIdx.x * K + kk;
    bf16_t* dst = &Bs[buf][threadIdx.x * 32];
#if USE_ASYNC_LDS
    b128_t* gp = (b128_t*)(uintptr_t)src;      // strip const, exact pointee type
    b128_t* lp = (b128_t*)(uintptr_t)dst;
    // INST_OFFSET applies to both global and LDS addresses (ISA 15.18.3)
    __builtin_amdgcn_global_load_async_to_lds_b128(gp, lp, 0, 0);
    __builtin_amdgcn_global_load_async_to_lds_b128(gp, lp, 16, 0);
    __builtin_amdgcn_global_load_async_to_lds_b128(gp, lp, 32, 0);
    __builtin_amdgcn_global_load_async_to_lds_b128(gp, lp, 48, 0);
#else
    ((v16bf*)dst)[0] = ((const v16bf*)src)[0];
    ((v16bf*)dst)[1] = ((const v16bf*)src)[1];
#endif
  };

  stage(0, 0);                                 // prologue: first tile in flight

  int cur = 0;
  for (int kk = 0; kk < K; kk += 32) {
#if USE_ASYNC_LDS
    __builtin_amdgcn_s_wait_asynccnt(0);       // tile issued last iteration has landed
#endif
    __syncthreads();                           // visible to all waves

    if (kk + 32 < K) stage(kk + 32, cur ^ 1);  // overlap next staging with compute

    // A fragment (16x32 bf16): lane<16 -> K 0-15 ; lane>=16 -> K 16-31
    union { v16bf v; v8bf h[2]; } af;
    const bf16_t* ap = A + (size_t)arow * K + kk + aklo;
    af.h[0] = *(const v8bf*)(ap);
    af.h[1] = *(const v8bf*)(ap + 8);
    if (kk + 32 < K) __builtin_prefetch(ap + 32, 0, 0);

    #pragma unroll
    for (int t = 0; t < 8; ++t) {
      // B fragment (32x16): lane col = l15, K base 0/16 -> contiguous 16 bf16 in N-major LDS
      const bf16_t* bp = &Bs[cur][(n0 + 16 * t + l15) * 32 + hi * 16];
      v16bf bf = *(const v16bf*)bp;
      acc[t] = __builtin_amdgcn_wmma_f32_16x16x32_bf16(
          false, af.v, false, bf, (short)0, acc[t], false, false);
    }
    __syncthreads();                           // all reads of buffer done
    cur ^= 1;
  }

  const int rbase = m0 + hi * 8;
  for (int t = 0; t < 8; ++t) {
    int col = n0 + 16 * t + l15;
    #pragma unroll
    for (int r = 0; r < 8; ++r) {
      int row = rbase + r;
      if (row < Nrows) Cf[(size_t)row * HC + col] = acc[t][r];
    }
  }
}

// ---------- attention prep: per (node, head) coefficients + init max/sum ----------
__global__ void k_attn_prep(const float* __restrict__ Hlin,
                            const float* __restrict__ a_s, const float* __restrict__ a_d,
                            float* __restrict__ es, float* __restrict__ ed,
                            int* __restrict__ nmax, float* __restrict__ z, int N) {
  int idx = blockIdx.x * blockDim.x + threadIdx.x;
  if (idx >= N * 4) return;
  int n = idx >> 2, h = idx & 3;
  const float* hp  = Hlin + (size_t)n * HC + h * CDIM;
  const float* asp = a_s + h * CDIM;
  const float* adp = a_d + h * CDIM;
  float s0 = 0.f, s1 = 0.f;
  for (int c = 0; c < CDIM; ++c) {
    float v = hp[c];
    s0 = fmaf(v, asp[c], s0);
    s1 = fmaf(v, adp[c], s1);
  }
  es[idx] = s0; ed[idx] = s1;
  nmax[idx] = INT_MIN; z[idx] = 0.f;
}

__global__ void k_init_out(float* __restrict__ Hout, const float* __restrict__ bg, int N) {
  int idx = blockIdx.x * blockDim.x + threadIdx.x;
  if (idx < N * HC) Hout[idx] = bg[idx & (HC - 1)];
}

// ---------- edge pass 1: leaky-relu scores + segment max (atomic on encoded int) ----------
__global__ void k_edge_max(const long long* __restrict__ ei,
                           const float* __restrict__ es, const float* __restrict__ ed,
                           float* __restrict__ esc, int* __restrict__ nmax, int E, int N) {
  int e = blockIdx.x * blockDim.x + threadIdx.x;
  if (e >= E + N) return;
  int s, d;
  if (e < E) { s = (int)ei[e]; d = (int)ei[(size_t)E + e]; }
  else       { s = e - E; d = s; }
  #pragma unroll
  for (int h = 0; h < 4; ++h) {
    float v = es[s * 4 + h] + ed[d * 4 + h];
    v = v > 0.f ? v : 0.2f * v;
    esc[(size_t)e * 4 + h] = v;
    atomicMax(&nmax[d * 4 + h], fenc(v));
  }
}

// ---------- edge pass 2: p = exp(e - max) + segment sum ----------
__global__ void k_edge_exp(const long long* __restrict__ ei,
                           float* __restrict__ esc, const int* __restrict__ nmax,
                           float* __restrict__ z, int E, int N) {
  int e = blockIdx.x * blockDim.x + threadIdx.x;
  if (e >= E + N) return;
  int d = (e < E) ? (int)ei[(size_t)E + e] : (e - E);
  #pragma unroll
  for (int h = 0; h < 4; ++h) {
    float p = __expf(esc[(size_t)e * 4 + h] - fdec(nmax[d * 4 + h]));
    esc[(size_t)e * 4 + h] = p;
    atomicAdd(&z[d * 4 + h], p);
  }
}

// ---------- edge pass 3: weighted message scatter; 64 threads/edge, float4 ----------
__global__ void k_edge_msg(const long long* __restrict__ ei,
                           const float* __restrict__ esc, const float* __restrict__ z,
                           const float* __restrict__ Hlin, float* __restrict__ Hout,
                           int E, int N) {
  int sub = threadIdx.x >> 6;          // edge slot within block (0..3)
  int t   = threadIdx.x & 63;          // 0..63 -> 4 floats each
  int e = blockIdx.x * 4 + sub;
  if (e >= E + N) return;
  int s, d;
  if (e < E) { s = (int)ei[e]; d = (int)ei[(size_t)E + e]; }
  else       { s = e - E; d = s; }
  int j = t << 2;                      // feature offset (0..252), 16B aligned
  int h = t >> 4;                      // head = j / 64
  float alpha = esc[(size_t)e * 4 + h] / (z[d * 4 + h] + 1e-16f);
  const float4 hv = *(const float4*)(Hlin + (size_t)s * HC + j);
  float* op = Hout + (size_t)d * HC + j;
  atomicAdd(op + 0, hv.x * alpha);
  atomicAdd(op + 1, hv.y * alpha);
  atomicAdd(op + 2, hv.z * alpha);
  atomicAdd(op + 3, hv.w * alpha);
}

// ---------- elu + convert to bf16 A (ld = 256), zero-pad rows >= N ----------
__global__ void k_elu_to_bf16(const float* __restrict__ Hout, bf16_t* __restrict__ A, int N) {
  size_t idx = (size_t)blockIdx.x * 256 + threadIdx.x;   // grid = Npad blocks
  int row = (int)(idx >> 8);
  float v = 0.f;
  if (row < N) v = elu1(Hout[idx]);
  A[idx] = (bf16_t)v;
}

// ---------- classifier: elu(h) @ Wc + bc, logits + softmax; one wave per node ----------
__global__ void k_classifier(const float* __restrict__ Hout, const float* __restrict__ Wc,
                             const float* __restrict__ bc, float* __restrict__ out, int N) {
  int wave = threadIdx.x >> 5, lane = threadIdx.x & 31;
  int n = blockIdx.x * 8 + wave;
  if (n >= N) return;
  float p0 = 0.f, p1 = 0.f;
  for (int j = lane; j < HC; j += 32) {
    float v = elu1(Hout[(size_t)n * HC + j]);
    p0 = fmaf(v, Wc[j * 2 + 0], p0);
    p1 = fmaf(v, Wc[j * 2 + 1], p1);
  }
  for (int off = 16; off; off >>= 1) {
    p0 += __shfl_xor(p0, off, 32);
    p1 += __shfl_xor(p1, off, 32);
  }
  if (lane == 0) {
    float l0 = p0 + bc[0], l1 = p1 + bc[1];
    out[(size_t)n * 2 + 0] = l0;
    out[(size_t)n * 2 + 1] = l1;
    float m = fmaxf(l0, l1);
    float e0 = __expf(l0 - m), e1 = __expf(l1 - m);
    float s = e0 + e1;
    out[(size_t)N * 2 + (size_t)n * 2 + 0] = e0 / s;
    out[(size_t)N * 2 + (size_t)n * 2 + 1] = e1 / s;
  }
}

extern "C" void kernel_launch(void* const* d_in, const int* in_sizes, int n_in,
                              void* d_out, int out_size, void* d_ws, size_t ws_size,
                              hipStream_t stream) {
  const float*     x   = (const float*)d_in[0];
  const long long* ei  = (const long long*)d_in[1];
  const float*     ets = (const float*)d_in[2];
  const float*     W1  = (const float*)d_in[3];
  const float*     b1  = (const float*)d_in[4];
  const float*     W2  = (const float*)d_in[5];
  const float*     b2  = (const float*)d_in[6];
  const float*     Wg1 = (const float*)d_in[7];
  const float*     as1 = (const float*)d_in[8];
  const float*     ad1 = (const float*)d_in[9];
  const float*     bg1 = (const float*)d_in[10];
  const float*     Wg2 = (const float*)d_in[11];
  const float*     as2 = (const float*)d_in[12];
  const float*     ad2 = (const float*)d_in[13];
  const float*     bg2 = (const float*)d_in[14];
  const float*     Wc  = (const float*)d_in[15];
  const float*     bc  = (const float*)d_in[16];
  float* out = (float*)d_out;

  const int N    = in_sizes[0] / INDIM;       // 50000
  const int E    = in_sizes[1] / 2;           // 800000
  const int Npad = ((N + 63) / 64) * 64;      // 50048
  const int Etot = E + N;                     // + self loops

  // carve workspace
  char* w = (char*)d_ws;
  size_t off = 0;
  auto carve = [&](size_t bytes) -> void* {
    void* p = w + off;
    off = (off + bytes + 255) & ~(size_t)255;
    return p;
  };
  bf16_t* A    = (bf16_t*)carve((size_t)Npad * KIN * sizeof(bf16_t));
  bf16_t* Bt1  = (bf16_t*)carve((size_t)HC * KIN * sizeof(bf16_t));
  bf16_t* Bt2  = (bf16_t*)carve((size_t)HC * HC * sizeof(bf16_t));
  float*  Hlin = (float*)carve((size_t)N * HC * sizeof(float));
  float*  Hout = (float*)carve((size_t)N * HC * sizeof(float));
  float*  es   = (float*)carve((size_t)N * 4 * sizeof(float));
  float*  ed   = (float*)carve((size_t)N * 4 * sizeof(float));
  int*    nmax = (int*)carve((size_t)N * 4 * sizeof(int));
  float*  z    = (float*)carve((size_t)N * 4 * sizeof(float));
  float*  esc  = (float*)carve((size_t)Etot * 4 * sizeof(float));

  // weight prep (bf16, N-major)
  k_wt_prep<<<(KIN * HC + 255) / 256, 256, 0, stream>>>(Wg1, Bt1, KIN, HC);
  k_wt_prep<<<(HC * HC + 255) / 256, 256, 0, stream>>>(Wg2, Bt2, HC, HC);

  // A0 = [x | temporal(ts[:N])] in bf16, padded
  k_build_A0<<<(N + 7) / 8, 256, 0, stream>>>(x, ets, W1, b1, W2, b2, A, N);
  {
    size_t cnt = (size_t)(Npad - N) * KIN;
    k_pad_bf16<<<(int)((cnt + 255) / 256), 256, 0, stream>>>(A, (size_t)N * KIN, cnt);
  }

  // ---- GAT layer 1 ----
  k_gemm_wmma<<<Npad / 64, 256, 0, stream>>>(A, Bt1, Hlin, KIN, N);
  k_attn_prep<<<(N * 4 + 255) / 256, 256, 0, stream>>>(Hlin, as1, ad1, es, ed, nmax, z, N);
  k_init_out<<<(N * HC + 255) / 256, 256, 0, stream>>>(Hout, bg1, N);
  k_edge_max<<<(Etot + 255) / 256, 256, 0, stream>>>(ei, es, ed, esc, nmax, E, N);
  k_edge_exp<<<(Etot + 255) / 256, 256, 0, stream>>>(ei, esc, nmax, z, E, N);
  k_edge_msg<<<(Etot + 3) / 4, 256, 0, stream>>>(ei, esc, z, Hlin, Hout, E, N);
  k_elu_to_bf16<<<Npad, 256, 0, stream>>>(Hout, A, N);   // A now ld=256

  // ---- GAT layer 2 ----
  k_gemm_wmma<<<Npad / 64, 256, 0, stream>>>(A, Bt2, Hlin, HC, N);
  k_attn_prep<<<(N * 4 + 255) / 256, 256, 0, stream>>>(Hlin, as2, ad2, es, ed, nmax, z, N);
  k_init_out<<<(N * HC + 255) / 256, 256, 0, stream>>>(Hout, bg2, N);
  k_edge_max<<<(Etot + 255) / 256, 256, 0, stream>>>(ei, es, ed, esc, nmax, E, N);
  k_edge_exp<<<(Etot + 255) / 256, 256, 0, stream>>>(ei, esc, nmax, z, E, N);
  k_edge_msg<<<(Etot + 3) / 4, 256, 0, stream>>>(ei, esc, z, Hlin, Hout, E, N);

  // ---- classifier + softmax ----
  k_classifier<<<(N + 7) / 8, 256, 0, stream>>>(Hout, Wc, bc, out, N);

  (void)n_in; (void)out_size; (void)ws_size;
}
